// ImageDiagnostic_55233279427342
// MI455X (gfx1250) — matrix operand order repaired and checked by reading, not compiled, not run on hardware
//
#include <hip/hip_runtime.h>

#define B_ 16
#define N_ 50000
#define M_ 128
#define EPS_ 1e-10f
#define SPLIT 25          // workgroups per batch along N
#define CHUNK 2000        // particles per workgroup (25*2000 = 50000 exactly)
#define KC 128            // particles per inner K-iteration
#define LOG2E_F 1.4426950408889634f

typedef _Float16 v16h __attribute__((ext_vector_type(16)));
typedef _Float16 v8h  __attribute__((ext_vector_type(8)));
typedef float    v8f  __attribute__((ext_vector_type(8)));

__global__ __launch_bounds__(256) void kde_zero(float* __restrict__ out) {
    int i = blockIdx.x * 256 + threadIdx.x;
    if (i < B_ * M_ * M_) out[i] = 0.0f;
}

__global__ __launch_bounds__(256) void kde_accum(
    const float* __restrict__ x, const float* __restrict__ y,
    const float* __restrict__ bins, const float* __restrict__ bw,
    float* __restrict__ out)
{
    // Double-buffered staging: ~71 KB LDS (CDNA5 allows up to 320 KB/WGP).
    __shared__ float xs[2][KC];
    __shared__ float ys[2][KC];
    // ky transposed: [y-bin N][particle K], row padded +8 halves so each row
    // stride is 272 B -> b128 reads across 16 rows hit 16 disjoint bank quads.
    __shared__ _Float16 kyT[2][M_][KC + 8];

    const int b    = blockIdx.y;
    const int s    = blockIdx.x;
    const int tid  = threadIdx.x;
    const int wave = tid >> 5;
    const int lane = tid & 31;
    const int lhalf = lane & 15;
    const int hi16  = lane >> 4;          // 0 or 1

    const float h  = bw[0];
    const float c2 = (-0.5f / (h * h)) * LOG2E_F;   // exp(c*d^2) == exp2(c2*d^2)

    // Fixed bin centers for this thread's roles.
    const float bA  = bins[wave * 16 + lhalf];   // A-matrix row (x-bin) for this wave
    const int   Nt  = tid & (M_ - 1);
    const float bN  = bins[Nt];                  // kyT row this thread fills
    const int   K0t = (tid >> 7) * (KC / 2);     // 0 or 64

    const int n_begin = s * CHUNK;
    const int n_stop  = min(n_begin + CHUNK, N_);
    const int nIters  = (n_stop - n_begin + KC - 1) / KC;

    v8f acc[8] = {};

    // ---- helpers ------------------------------------------------------
    auto stage = [&](int it, int buf) {
        const int idx = tid & (KC - 1);
        const int n = n_begin + it * KC + idx;
        const float* src = (tid < KC) ? x : y;
        const float v = (n < n_stop) ? src[b * N_ + n] : 1e20f;  // tail -> exp==0
        if (tid < KC) xs[buf][idx] = v; else ys[buf][idx] = v;
    };
    auto kyfill = [&](int buf) {
        #pragma unroll
        for (int kk = 0; kk < 8; ++kk) {
            v8h v;
            #pragma unroll
            for (int hh = 0; hh < 8; ++hh) {
                const float d = ys[buf][K0t + kk * 8 + hh] - bN;
                v[hh] = (_Float16)__builtin_amdgcn_exp2f(c2 * d * d);
            }
            *(v8h*)&kyT[buf][Nt][K0t + kk * 8] = v;
        }
    };

    // ---- prologue: stage + ky-fill chunk 0 ---------------------------
    stage(0, 0);
    __syncthreads();
    kyfill(0);

    for (int i = 0; i < nIters; ++i) {
        const int buf = i & 1;
        __syncthreads();                      // kyfill(i) visible; prev WMMA reads done
        if (i + 1 < nIters) stage(i + 1, buf ^ 1);
        __syncthreads();                      // ys[buf^1] staged
        if (i + 1 < nIters) kyfill(buf ^ 1);  // overlaps with WMMA chain below

        // A fragments for all 4 K-steps (ISA 7.12.2 16-bit A layout:
        // lane>=16 => K-base +8, halves 8..15 => K +16).
        v16h afrag[4];
        #pragma unroll
        for (int kk = 0; kk < 4; ++kk) {
            const int kbase = kk * 32 + hi16 * 8;
            #pragma unroll
            for (int hh = 0; hh < 16; ++hh) {
                const int K = kbase + ((hh < 8) ? hh : (hh + 8));
                const float d = xs[buf][K] - bA;
                afrag[kk][hh] = (_Float16)__builtin_amdgcn_exp2f(c2 * d * d);
            }
        }

        // WMMA chain with double-buffered B-fragment LDS loads.
        // B 32x16 f16 layout: lanes 0-15 K=0..7,16..23, lanes 16-31 K=8..15,24..31,
        // N = lane%16 -> two 16B chunks per lane in a kyT row.
        v8h blo0, bhi0, blo1, bhi1;
        {
            const int Ncol = lhalf;            // idx 0: kk=0, t=0
            const int Kb   = hi16 * 8;
            blo0 = *(const v8h*)&kyT[buf][Ncol][Kb];
            bhi0 = *(const v8h*)&kyT[buf][Ncol][Kb + 16];
        }
        #pragma unroll
        for (int idx = 0; idx < 32; ++idx) {
            const int kk = idx >> 3;
            const int t  = idx & 7;
            if (idx + 1 < 32) {                // prefetch next tile's fragment
                const int nk = (idx + 1) >> 3;
                const int nt = (idx + 1) & 7;
                const int Ncol = nt * 16 + lhalf;
                const int Kb   = nk * 32 + hi16 * 8;
                if (idx & 1) {
                    blo0 = *(const v8h*)&kyT[buf][Ncol][Kb];
                    bhi0 = *(const v8h*)&kyT[buf][Ncol][Kb + 16];
                } else {
                    blo1 = *(const v8h*)&kyT[buf][Ncol][Kb];
                    bhi1 = *(const v8h*)&kyT[buf][Ncol][Kb + 16];
                }
            }
            const v8h lo = (idx & 1) ? blo1 : blo0;
            const v8h hi = (idx & 1) ? bhi1 : bhi0;
            const v16h bm = __builtin_shufflevector(
                lo, hi, 0, 1, 2, 3, 4, 5, 6, 7, 8, 9, 10, 11, 12, 13, 14, 15);
            acc[t] = __builtin_amdgcn_wmma_f32_16x16x32_f16(
                false, afrag[kk], false, bm, (short)0, acc[t], false, false);
        }
    }

    // ---- combine N-splits: C/D layout VGPR r -> row r (+8 for lanes>=16) ----
    #pragma unroll
    for (int t = 0; t < 8; ++t) {
        #pragma unroll
        for (int r = 0; r < 8; ++r) {
            const int m   = wave * 16 + r + hi16 * 8;
            const int col = t * 16 + lhalf;
            __hip_atomic_fetch_add(&out[(b * M_ + m) * M_ + col], acc[t][r],
                                   __ATOMIC_RELAXED, __HIP_MEMORY_SCOPE_AGENT);
        }
    }
}

__global__ __launch_bounds__(256) void kde_norm(float* __restrict__ out) {
    __shared__ float red[256];
    const int b = blockIdx.x;
    const int tid = threadIdx.x;
    float s = 0.0f;
    for (int i = tid; i < M_ * M_; i += 256) s += out[b * M_ * M_ + i];
    red[tid] = s;
    __syncthreads();
    for (int off = 128; off > 0; off >>= 1) {
        if (tid < off) red[tid] += red[tid + off];
        __syncthreads();
    }
    const float inv = 1.0f / (red[0] + EPS_);
    for (int i = tid; i < M_ * M_; i += 256) out[b * M_ * M_ + i] *= inv;
}

extern "C" void kernel_launch(void* const* d_in, const int* in_sizes, int n_in,
                              void* d_out, int out_size, void* d_ws, size_t ws_size,
                              hipStream_t stream) {
    const float* x    = (const float*)d_in[0];
    const float* y    = (const float*)d_in[1];
    const float* bins = (const float*)d_in[2];
    const float* bw   = (const float*)d_in[3];
    float* out = (float*)d_out;

    (void)d_ws; (void)ws_size; (void)in_sizes; (void)n_in; (void)out_size;

    kde_zero<<<(B_ * M_ * M_ + 255) / 256, 256, 0, stream>>>(out);
    kde_accum<<<dim3(SPLIT, B_), 256, 0, stream>>>(x, y, bins, bw, out);
    kde_norm<<<B_, 256, 0, stream>>>(out);
}
